// SpectrumComposer_26912265077242
// MI455X (gfx1250) — matrix operand order repaired and checked by reading, not compiled, not run on hardware
//
#include <hip/hip_runtime.h>

typedef __attribute__((ext_vector_type(16))) _Float16 v16h;
typedef __attribute__((ext_vector_type(8)))  float    v8f;

#define HDIM 256      // H = SIZE_IN/2
#define SIN_ 512      // SIZE_IN (channels)
#define SOUT 512      // SIZE_OUT (n dimension)
#define KDIM 512      // GEMM K = 2*H
#define MDIM 2048     // 4 * 512 rows

// ---------------------------------------------------------------------------
// A[m][h]      = amp[m][h] * sin(phase[m][h])          (h in [0,256))
// A[m][256+h]  = amp[m][h] * cos(phase[m][h])
// Row-major f16: per-lane fragment dwords form two contiguous 16B runs.
// ---------------------------------------------------------------------------
__global__ __launch_bounds__(256) void a_prep(const float* __restrict__ in,
                                              _Float16* __restrict__ A) {
    int idx = blockIdx.x * blockDim.x + threadIdx.x;   // 0 .. 2048*256
    if (idx >= MDIM * HDIM) return;
    int m = idx >> 8;          // row (b*s)
    int h = idx & 255;
    float ph = in[m * SIN_ + h];
    float am = in[m * SIN_ + HDIM + h];
    float s, c;
    sincosf(ph, &s, &c);
    A[m * KDIM + h]        = (_Float16)(am * s);
    A[m * KDIM + HDIM + h] = (_Float16)(am * c);
}

// ---------------------------------------------------------------------------
// B[k][n] = cos(w[k]   * t[n])  for k in [0,256)
// B[k][n] = sin(w[k-256]*t[n])  for k in [256,512)
// Stored fragment-major: for K-pair p (= {f16 B[2p][n], B[2p+1][n]} packed in
// one dword) the dword lives at Bf[((p>>4)*SOUT + n)*16 + (p&15)].
// => for a given K-block and lane, the 8 fragment dwords form two 16B runs
//    (each 16B-aligned) -> two global_load_b128 per B fragment in the GEMM.
// ---------------------------------------------------------------------------
__global__ __launch_bounds__(256) void b_prep(const float* __restrict__ w,
                                              unsigned int* __restrict__ Bf) {
    int idx = blockIdx.x * blockDim.x + threadIdx.x;   // 0 .. 256*512
    if (idx >= (KDIM / 2) * SOUT) return;
    int p = idx >> 9;          // K-pair index 0..255
    int n = idx & 511;
    float t  = (float)n / (float)(SOUT - 1);
    int k0 = 2 * p;
    float w0 = w[k0 & (HDIM - 1)];
    float w1 = w[(k0 + 1) & (HDIM - 1)];
    float v0, v1;
    if (k0 < HDIM) { v0 = cosf(w0 * t); v1 = cosf(w1 * t); }
    else           { v0 = sinf(w0 * t); v1 = sinf(w1 * t); }
    union { _Float16 h[2]; unsigned int u; } pk;
    pk.h[0] = (_Float16)v0;
    pk.h[1] = (_Float16)v1;
    Bf[(((unsigned)p >> 4) * SOUT + n) * 16 + (p & 15)] = pk.u;
}

// ---------------------------------------------------------------------------
// C[m][n] = sum_k A[m][k] * B[k][n].
// One wave computes a 32x32 C patch = 2x2 WMMA tiles (4 independent f32
// accumulators). Inner loop: 4 A-b128 + 4 B-b128 loads feeding 4 WMMAs.
// Fragments follow the CDNA5 16-bit A / B and 32-bit C VGPR layouts (wave32).
// ---------------------------------------------------------------------------
__global__ __launch_bounds__(256) void wmma_gemm(const _Float16* __restrict__ A,
                                                 const unsigned int* __restrict__ Bf,
                                                 float* __restrict__ C) {
    int wave  = threadIdx.x >> 5;
    int lane  = threadIdx.x & 31;
    int task  = blockIdx.x * 8 + wave;     // 1024 tasks total
    int mt    = task >> 4;                 // 64 m-patches (32 rows each)
    int nt    = task & 15;                 // 16 n-patches (32 cols each)
    int m0    = mt << 5;
    int n0    = nt << 5;
    int lhalf = lane >> 4;                 // 0: lanes 0-15, 1: lanes 16-31
    int l15   = lane & 15;

    union Frag { v16h v; unsigned int u[8]; uint4 q[2]; };
    Frag a0, a1, b0, b1;
    v8f acc00 = {}, acc01 = {}, acc10 = {}, acc11 = {};

    // A rows for this lane, viewed as packed f16 pairs (dwords).
    // Lane fragment for K-block kidx: dwords [16*kidx + 4*lhalf + 0..3] and
    // [16*kidx + 8 + 4*lhalf + 0..3]  (lanes 16-31 = same rows, K shifted +8).
    const unsigned int* Arow0 = (const unsigned int*)(A + (m0 + l15) * KDIM)
                                + 4 * lhalf;
    const unsigned int* Arow1 = Arow0 + 16 * (KDIM / 2);   // +16 rows
    // B fragment base: pair p = kidx*16 + 4*lhalf + j (j<4), +8 for j>=4, column n.
    const unsigned int* Bc0 = Bf + (n0 + l15) * 16 + 4 * lhalf;
    const unsigned int* Bc1 = Bc0 + 16 * 16;               // +16 columns

    for (int kidx = 0; kidx < KDIM / 32; ++kidx) {
        int adw = kidx * 16;
        a0.q[0] = *(const uint4*)(Arow0 + adw);
        a0.q[1] = *(const uint4*)(Arow0 + adw + 8);
        a1.q[0] = *(const uint4*)(Arow1 + adw);
        a1.q[1] = *(const uint4*)(Arow1 + adw + 8);

        int bdw = kidx * SOUT * 16;
        b0.q[0] = *(const uint4*)(Bc0 + bdw);
        b0.q[1] = *(const uint4*)(Bc0 + bdw + 8);
        b1.q[0] = *(const uint4*)(Bc1 + bdw);
        b1.q[1] = *(const uint4*)(Bc1 + bdw + 8);

        // 4 independent WMMAs -> no D->A/B hazards, back-to-back issue.
        acc00 = __builtin_amdgcn_wmma_f32_16x16x32_f16(false, a0.v, false, b0.v,
                                                       (short)0, acc00, false, false);
        acc01 = __builtin_amdgcn_wmma_f32_16x16x32_f16(false, a0.v, false, b1.v,
                                                       (short)0, acc01, false, false);
        acc10 = __builtin_amdgcn_wmma_f32_16x16x32_f16(false, a1.v, false, b0.v,
                                                       (short)0, acc10, false, false);
        acc11 = __builtin_amdgcn_wmma_f32_16x16x32_f16(false, a1.v, false, b1.v,
                                                       (short)0, acc11, false, false);
    }

    // C 16x16 f32 layout: VGPR r -> M = r + 8*lhalf, N = base + (lane&15).
    int n = n0 + l15;
#pragma unroll
    for (int r = 0; r < 8; ++r) {
        int row = m0 + r + 8 * lhalf;
        C[row * SOUT + n]             = acc00[r];
        C[row * SOUT + n + 16]        = acc01[r];
        C[(row + 16) * SOUT + n]      = acc10[r];
        C[(row + 16) * SOUT + n + 16] = acc11[r];
    }
}

extern "C" void kernel_launch(void* const* d_in, const int* in_sizes, int n_in,
                              void* d_out, int out_size, void* d_ws, size_t ws_size,
                              hipStream_t stream) {
    (void)in_sizes; (void)n_in; (void)out_size; (void)ws_size;
    const float* in = (const float*)d_in[0];   // (4,512,512) f32
    const float* w  = (const float*)d_in[1];   // (256,) f32
    float* out = (float*)d_out;                // (4,512,512) f32

    _Float16*     A  = (_Float16*)d_ws;                                   // 2   MB
    unsigned int* Bf = (unsigned int*)((char*)d_ws +
                        (size_t)MDIM * KDIM * sizeof(_Float16));          // 0.5 MB

    a_prep<<<(MDIM * HDIM) / 256, 256, 0, stream>>>(in, A);
    b_prep<<<((KDIM / 2) * SOUT) / 256, 256, 0, stream>>>(w, Bf);
    wmma_gemm<<<(MDIM / 32) * (SOUT / 32) / 8, 256, 0, stream>>>(A, Bf, out);
}
